// DynamicMemoryInduction_12713103197271
// MI455X (gfx1250) — matrix operand order repaired and checked by reading, not compiled, not run on hardware
//
#include <hip/hip_runtime.h>
#include <cstdint>
#include <cstddef>

// Problem constants (match reference)
#define I_CAPS 512
#define QN     512
#define CAPS   64
#define DIM    16
#define KDIM   768
#define NDIM   1024   // CAPS*DIM

typedef __attribute__((ext_vector_type(16))) __bf16 v16bf;
typedef __attribute__((ext_vector_type(8)))  __bf16 v8bf;
typedef __attribute__((ext_vector_type(8)))  float  v8f;

__device__ __forceinline__ v8bf ld8(const __bf16* p) { return *(const v8bf*)p; }
__device__ __forceinline__ v16bf cat16(v8bf a, v8bf b) {
  return __builtin_shufflevector(a, b, 0,1,2,3,4,5,6,7,8,9,10,11,12,13,14,15);
}
__device__ __forceinline__ v8f wmma_bf(v16bf a, v16bf b, v8f c) {
  // D = A(16x32 bf16) * B(32x16 bf16) + C(16x16 f32)
  return __builtin_amdgcn_wmma_f32_16x16x32_bf16(false, a, false, b, (short)0, c, false, false);
}

// ---------------------------------------------------------------- converts
__global__ void k_f32_to_bf16(const float* __restrict__ in, __bf16* __restrict__ out, int n) {
  int i = blockIdx.x * blockDim.x + threadIdx.x;
  if (i < n) out[i] = (__bf16)in[i];
}

// W[768][1024] f32 -> Wt[1024][768] bf16 (column of W contiguous for B-operand loads)
__global__ void k_transpose_w(const float* __restrict__ W, __bf16* __restrict__ Wt) {
  int t = blockIdx.x * blockDim.x + threadIdx.x;   // k*1024 + n
  int k = t >> 10, n = t & 1023;
  Wt[n * KDIM + k] = (__bf16)W[t];
}

// ------------------------------------------------- projection GEMM (bf16 WMMA)
// out layout: [c][row][d] f32  (c = n>>4, d = n&15), c-stride = M*16
__global__ __launch_bounds__(256)
void k_proj_gemm(const __bf16* __restrict__ A,   // [M][768] row-major bf16
                 const __bf16* __restrict__ Bt,  // [1024][768] (col of W contiguous)
                 const float*  __restrict__ bias,// [1024]
                 float* __restrict__ out, int M) {
  const int lane = threadIdx.x & 31;
  const int wave = threadIdx.x >> 5;
  const int l16 = lane & 15, hi = lane >> 4;
  const int mBase = blockIdx.x * 128 + (wave & 3) * 32;
  const int nBase = blockIdx.y * 64  + (wave >> 2) * 32;

  const __bf16* aR0 = A  + (size_t)(mBase + l16)      * KDIM;
  const __bf16* aR1 = A  + (size_t)(mBase + 16 + l16) * KDIM;
  const __bf16* bR0 = Bt + (size_t)(nBase + l16)      * KDIM;
  const __bf16* bR1 = Bt + (size_t)(nBase + 16 + l16) * KDIM;

  v8f acc[2][2] = {};
  for (int k0 = 0; k0 < KDIM; k0 += 32) {
    const int ka = k0 + hi * 8;    // A: lanes>=16 start at K+8
    const int kb = k0 + hi * 8;    // B: lanes>=16 start at K+8
    v16bf a0 = cat16(ld8(aR0 + ka), ld8(aR0 + ka + 16));
    v16bf a1 = cat16(ld8(aR1 + ka), ld8(aR1 + ka + 16));
    v16bf b0 = cat16(ld8(bR0 + kb), ld8(bR0 + kb + 16));
    v16bf b1 = cat16(ld8(bR1 + kb), ld8(bR1 + kb + 16));
    acc[0][0] = wmma_bf(a0, b0, acc[0][0]);
    acc[0][1] = wmma_bf(a0, b1, acc[0][1]);
    acc[1][0] = wmma_bf(a1, b0, acc[1][0]);
    acc[1][1] = wmma_bf(a1, b1, acc[1][1]);
  }
  #pragma unroll
  for (int mi = 0; mi < 2; ++mi)
    #pragma unroll
    for (int ni = 0; ni < 2; ++ni) {
      int n = nBase + ni * 16 + l16;
      float b = bias[n];
      size_t cbase = (size_t)(n >> 4) * M * 16 + (n & 15);
      #pragma unroll
      for (int r = 0; r < 8; ++r) {
        int row = mBase + mi * 16 + hi * 8 + r;
        out[cbase + (size_t)row * 16] = acc[mi][ni][r] + b;
      }
    }
}

// --------------------------------------------- prepare m-side routing operands
// hatm [c][i][16] f32 -> msg (raw bf16 [c][i][d]), msgT (raw bf16 [c][d][i]),
//                        xm (centered bf16 [c][i][d]), sxx f32 [c][i]
__global__ void k_prep_m(const float* __restrict__ hatm, __bf16* __restrict__ msg,
                         __bf16* __restrict__ msgT, __bf16* __restrict__ xm,
                         float* __restrict__ sxx) {
  int t = blockIdx.x * blockDim.x + threadIdx.x;   // c*512 + i
  int c = t >> 9, i = t & 511;
  size_t base = (size_t)t * 16;
  float s[16], mean = 0.f;
  #pragma unroll
  for (int d = 0; d < 16; ++d) { s[d] = hatm[base + d]; mean += s[d]; }
  mean *= (1.0f / 16.0f);
  float ss = 0.f;
  #pragma unroll
  for (int d = 0; d < 16; ++d) {
    float xc = s[d] - mean;
    msg[base + d]  = (__bf16)s[d];
    xm[base + d]   = (__bf16)xc;
    msgT[((size_t)c * 16 + d) * 512 + i] = (__bf16)s[d];
    ss += xc * xc;
  }
  sxx[t] = ss;
}

// qcur [c][q][16] f32 -> ym centered bf16 [c][q][d], syy f32 [c][q]
__global__ void k_prep_q(const float* __restrict__ qcur, __bf16* __restrict__ ym,
                         float* __restrict__ syy) {
  int t = blockIdx.x * blockDim.x + threadIdx.x;   // c*512 + q
  size_t base = (size_t)t * 16;
  float s[16], mean = 0.f;
  #pragma unroll
  for (int d = 0; d < 16; ++d) { s[d] = qcur[base + d]; mean += s[d]; }
  mean *= (1.0f / 16.0f);
  float ss = 0.f;
  #pragma unroll
  for (int d = 0; d < 16; ++d) {
    float yc = s[d] - mean;
    ym[base + d] = (__bf16)yc;
    ss += yc * yc;
  }
  syy[t] = ss;
}

// --------------------------- p = tanh(pearson): one zero-padded WMMA per tile
__global__ __launch_bounds__(256)
void k_pearson(const __bf16* __restrict__ ym, const __bf16* __restrict__ xm,
               const float* __restrict__ sxx, const float* __restrict__ syy,
               __bf16* __restrict__ p) {
  int wave = (blockIdx.x * blockDim.x + threadIdx.x) >> 5;  // c*1024 + qt*32 + it
  int lane = threadIdx.x & 31;
  int it = wave & 31, qt = (wave >> 5) & 31, c = wave >> 10;
  int l16 = lane & 15, hi = lane >> 4;
  int qBase = qt * 16, iBase = it * 16;
  v8bf z8 = {};
  // A: 16(q) x 32(K=d, upper 16 zero)
  v16bf av = cat16(ld8(ym + ((size_t)c * 512 + qBase + l16) * 16 + hi * 8), z8);
  // B: 32(K=d) x 16(i); lanes>=16 hold K=8..15, upper halves (K=16..31) zero
  const __bf16* xr = xm + ((size_t)c * 512 + iBase + l16) * 16;
  v16bf bv = cat16(ld8(xr + hi * 8), z8);
  v8f acc = {};
  acc = wmma_bf(av, bv, acc);
  float sx = sxx[c * 512 + iBase + l16];
  #pragma unroll
  for (int r = 0; r < 8; ++r) {
    int q = qBase + hi * 8 + r;
    float den = sx * syy[c * 512 + q] + 1e-8f;
    float val = tanhf(acc[r] * rsqrtf(den));
    p[((size_t)c * 512 + q) * 512 + iBase + l16] = (__bf16)val;
  }
}

// -------------------- dsp = softmax_c(a) + p   (a==0 on first iteration)
__global__ __launch_bounds__(256)
void k_dsp(const float* __restrict__ a, const __bf16* __restrict__ p,
           __bf16* __restrict__ dsp, int hasA) {
  size_t qi = (size_t)blockIdx.x * blockDim.x + threadIdx.x;  // q*512 + i
  const size_t QI = (size_t)QN * I_CAPS;
  if (hasA) {
    float av[64]; float mx = -3.4e38f;
    #pragma unroll
    for (int c = 0; c < 64; ++c) { av[c] = a[c * QI + qi]; mx = fmaxf(mx, av[c]); }
    float s = 0.f;
    #pragma unroll
    for (int c = 0; c < 64; ++c) { av[c] = __expf(av[c] - mx); s += av[c]; }
    float inv = 1.0f / s;
    #pragma unroll
    for (int c = 0; c < 64; ++c)
      dsp[c * QI + qi] = (__bf16)(av[c] * inv + (float)p[c * QI + qi]);
  } else {
    #pragma unroll
    for (int c = 0; c < 64; ++c)
      dsp[c * QI + qi] = (__bf16)((1.0f / 64.0f) + (float)p[c * QI + qi]);
  }
}

// -------------------- hat_v[c][q][d] = sum_i dsp[c][q][i] * msgT[c][d][i]  (K=512)
__global__ __launch_bounds__(256)
void k_vgemm(const __bf16* __restrict__ dsp, const __bf16* __restrict__ msgT,
             float* __restrict__ hat_v) {
  int wave = (blockIdx.x * blockDim.x + threadIdx.x) >> 5;  // c*32 + qt
  int lane = threadIdx.x & 31;
  int l16 = lane & 15, hi = lane >> 4;
  int qt = wave & 31, c = wave >> 5;
  int qBase = qt * 16;
  const __bf16* ar = dsp  + ((size_t)c * 512 + qBase + l16) * 512;
  const __bf16* br = msgT + ((size_t)c * 16 + l16) * 512;   // column d = l16
  v8f acc = {};
  for (int i0 = 0; i0 < 512; i0 += 32) {
    v16bf av = cat16(ld8(ar + i0 + hi * 8),  ld8(ar + i0 + hi * 8 + 16));
    v16bf bv = cat16(ld8(br + i0 + hi * 8), ld8(br + i0 + hi * 8 + 16));
    acc = wmma_bf(av, bv, acc);
  }
  #pragma unroll
  for (int r = 0; r < 8; ++r)
    hat_v[((size_t)c * 512 + qBase + hi * 8 + r) * 16 + l16] = acc[r];
}

// ------------- v = squash(hat_v); qcur = (qcur+v)/2; ym/syy of new qcur
__global__ void k_postv(const float* __restrict__ hat_v, float* __restrict__ qcur,
                        __bf16* __restrict__ v_bf, __bf16* __restrict__ ym,
                        float* __restrict__ syy) {
  int t = blockIdx.x * blockDim.x + threadIdx.x;   // c*512 + q
  size_t base = (size_t)t * 16;
  float s[16], n2 = 0.f;
  #pragma unroll
  for (int d = 0; d < 16; ++d) { s[d] = hat_v[base + d]; n2 += s[d] * s[d]; }
  float scale = n2 / ((1.0f + n2) * sqrtf(n2 + 1e-8f));
  float qn[16], mean = 0.f;
  #pragma unroll
  for (int d = 0; d < 16; ++d) {
    float v = s[d] * scale;
    v_bf[base + d] = (__bf16)v;
    float qq = (qcur[base + d] + v) * 0.5f;
    qcur[base + d] = qq;
    qn[d] = qq; mean += qq;
  }
  mean *= (1.0f / 16.0f);
  float ss = 0.f;
  #pragma unroll
  for (int d = 0; d < 16; ++d) {
    float yc = qn[d] - mean;
    ym[base + d] = (__bf16)yc;
    ss += yc * yc;
  }
  syy[t] = ss;
}

// ---------- cos = v . m ; a += p * cos  (one padded WMMA per 16x16 tile)
__global__ __launch_bounds__(256)
void k_cos(const __bf16* __restrict__ v_bf, const __bf16* __restrict__ msg,
           const __bf16* __restrict__ p, float* __restrict__ a, int hasA) {
  int wave = (blockIdx.x * blockDim.x + threadIdx.x) >> 5;
  int lane = threadIdx.x & 31;
  int it = wave & 31, qt = (wave >> 5) & 31, c = wave >> 10;
  int l16 = lane & 15, hi = lane >> 4;
  int qBase = qt * 16, iBase = it * 16;
  v8bf z8 = {};
  v16bf av = cat16(ld8(v_bf + ((size_t)c * 512 + qBase + l16) * 16 + hi * 8), z8);
  const __bf16* mr = msg + ((size_t)c * 512 + iBase + l16) * 16;
  v16bf bv = cat16(ld8(mr + hi * 8), z8);
  v8f acc = {};
  acc = wmma_bf(av, bv, acc);
  #pragma unroll
  for (int r = 0; r < 8; ++r) {
    int q = qBase + hi * 8 + r;
    size_t idx = ((size_t)c * 512 + q) * 512 + iBase + l16;
    float upd = (float)p[idx] * acc[r];
    a[idx] = hasA ? (a[idx] + upd) : upd;
  }
}

// ---------- final: out[q][c*16+d] = squash(hat_v)[c][q][d]
__global__ void k_final(const float* __restrict__ hat_v, float* __restrict__ out) {
  int t = blockIdx.x * blockDim.x + threadIdx.x;   // c*512 + q
  int c = t >> 9, q = t & 511;
  size_t base = (size_t)t * 16;
  float s[16], n2 = 0.f;
  #pragma unroll
  for (int d = 0; d < 16; ++d) { s[d] = hat_v[base + d]; n2 += s[d] * s[d]; }
  float scale = n2 / ((1.0f + n2) * sqrtf(n2 + 1e-8f));
  #pragma unroll
  for (int d = 0; d < 16; ++d)
    out[(size_t)q * 1024 + c * 16 + d] = s[d] * scale;
}

// ================================================================ launcher
extern "C" void kernel_launch(void* const* d_in, const int* in_sizes, int n_in,
                              void* d_out, int out_size, void* d_ws, size_t ws_size,
                              hipStream_t stream) {
  const float* m   = (const float*)d_in[0];   // [512][768]
  const float* q   = (const float*)d_in[1];   // [512][768]
  const float* W_w = (const float*)d_in[2];   // [768][1024]
  const float* W_b = (const float*)d_in[3];   // [1024]
  float* out = (float*)d_out;                 // [512][1024]

  // bump allocator over workspace (~150 MB total)
  char* w = (char*)d_ws; size_t off = 0;
  auto alloc = [&](size_t bytes) -> void* {
    void* ptr = w + off;
    off = (off + bytes + 255) & ~(size_t)255;
    return ptr;
  };
  const size_t QI = (size_t)QN * I_CAPS;                 // 262144
  __bf16* mbf   = (__bf16*)alloc((size_t)I_CAPS * KDIM * 2);
  __bf16* qbf   = (__bf16*)alloc((size_t)QN * KDIM * 2);
  __bf16* wtbf  = (__bf16*)alloc((size_t)NDIM * KDIM * 2);
  float*  hatm  = (float*) alloc((size_t)CAPS * I_CAPS * DIM * 4);  // [c][i][d]
  float*  qcur  = (float*) alloc((size_t)CAPS * QN * DIM * 4);      // [c][q][d]
  __bf16* msg   = (__bf16*)alloc((size_t)CAPS * I_CAPS * DIM * 2);
  __bf16* msgT  = (__bf16*)alloc((size_t)CAPS * DIM * I_CAPS * 2);
  __bf16* xm    = (__bf16*)alloc((size_t)CAPS * I_CAPS * DIM * 2);
  float*  sxx   = (float*) alloc((size_t)CAPS * I_CAPS * 4);
  __bf16* ym    = (__bf16*)alloc((size_t)CAPS * QN * DIM * 2);
  float*  syy   = (float*) alloc((size_t)CAPS * QN * 4);
  __bf16* v_bf  = (__bf16*)alloc((size_t)CAPS * QN * DIM * 2);
  __bf16* p_bf  = (__bf16*)alloc(CAPS * QI * 2);                    // 33 MB
  __bf16* dspb  = (__bf16*)alloc(CAPS * QI * 2);                    // 33 MB
  float*  a_log = (float*) alloc(CAPS * QI * 4);                    // 67 MB
  float*  hat_v = (float*) alloc((size_t)CAPS * QN * DIM * 4);
  (void)ws_size; (void)in_sizes; (void)n_in; (void)out_size;

  // 1) bf16 packs
  k_f32_to_bf16<<<(I_CAPS * KDIM + 255) / 256, 256, 0, stream>>>(m, mbf, I_CAPS * KDIM);
  k_f32_to_bf16<<<(QN * KDIM + 255) / 256, 256, 0, stream>>>(q, qbf, QN * KDIM);
  k_transpose_w<<<(KDIM * NDIM) / 256, 256, 0, stream>>>(W_w, wtbf);

  // 2) projections (WMMA): hat_m -> [c][i][d], hat_q -> qcur [c][q][d]
  k_proj_gemm<<<dim3(4, 16), 256, 0, stream>>>(mbf, wtbf, W_b, hatm, I_CAPS);
  k_proj_gemm<<<dim3(4, 16), 256, 0, stream>>>(qbf, wtbf, W_b, qcur, QN);

  // 3) routing operand prep
  k_prep_m<<<(CAPS * I_CAPS) / 256, 256, 0, stream>>>(hatm, msg, msgT, xm, sxx);
  k_prep_q<<<(CAPS * QN) / 256, 256, 0, stream>>>(qcur, ym, syy);

  // 4) p0
  k_pearson<<<(CAPS * 32 * 32) / 8, 256, 0, stream>>>(ym, xm, sxx, syy, p_bf);

  // 5) two routing iterations
  for (int t = 0; t < 2; ++t) {
    k_dsp<<<(int)(QI / 256), 256, 0, stream>>>(a_log, p_bf, dspb, t > 0);
    k_vgemm<<<(CAPS * 32) / 8, 256, 0, stream>>>(dspb, msgT, hat_v);
    k_postv<<<(CAPS * QN) / 256, 256, 0, stream>>>(hat_v, qcur, v_bf, ym, syy);
    k_cos<<<(CAPS * 32 * 32) / 8, 256, 0, stream>>>(v_bf, msg, p_bf, a_log, t > 0);
    k_pearson<<<(CAPS * 32 * 32) / 8, 256, 0, stream>>>(ym, xm, sxx, syy, p_bf);
  }

  // 6) final aggregation + squash -> output
  k_dsp<<<(int)(QI / 256), 256, 0, stream>>>(a_log, p_bf, dspb, 1);
  k_vgemm<<<(CAPS * 32) / 8, 256, 0, stream>>>(dspb, msgT, hat_v);
  k_final<<<(CAPS * QN) / 256, 256, 0, stream>>>(hat_v, out);
}
